// MambaLMHeadModelSafe_24017457119638
// MI455X (gfx1250) — matrix operand order repaired and checked by reading, not compiled, not run on hardware
//
#include <hip/hip_runtime.h>

// ---------------------------------------------------------------------------
// Mamba forward for MI455X (gfx1250, wave32).
// GEMMs: bf16 V_WMMA_F32_16X16X32_BF16, f32 accumulate.
// Scan:  register-resident per-channel recurrence (16 states/thread).
// ---------------------------------------------------------------------------

typedef __bf16 bf16_t;
typedef __attribute__((ext_vector_type(16))) __bf16 v16bf;
typedef __attribute__((ext_vector_type(8)))  __bf16 v8bf;
typedef __attribute__((ext_vector_type(8)))  float  v8f;

union BFrag { v16bf v; v8bf h[2]; };

__device__ __forceinline__ bf16_t f2bf(float f) {
    union { float f; unsigned u; } in; in.f = f;
    unsigned u = in.u;
    unsigned r = u + 0x7FFFu + ((u >> 16) & 1u);      // round-to-nearest-even
    union { unsigned short s; bf16_t b; } out;
    out.s = (unsigned short)(r >> 16);
    return out.b;
}

__device__ __forceinline__ float silu_f(float x) {
    return x / (1.0f + __expf(-x));
}

// --------------------------- f32 -> bf16 convert ---------------------------
__global__ void f32_to_bf16_kernel(const float* __restrict__ src,
                                   bf16_t* __restrict__ dst, long n) {
    long i = (long)blockIdx.x * blockDim.x + threadIdx.x;
    long stride = (long)gridDim.x * blockDim.x;
    for (; i < n; i += stride) dst[i] = f2bf(src[i]);
}

// ------------------------------ embedding gather ---------------------------
__global__ void embed_kernel(const int* __restrict__ ids,
                             const float* __restrict__ emb,
                             float* __restrict__ hidden, int L, int D) {
    long i = (long)blockIdx.x * blockDim.x + threadIdx.x;
    if (i >= (long)L * D) return;
    int l = (int)(i / D), j = (int)(i % D);
    hidden[i] = emb[(long)ids[l] * D + j];
}

// ------------------- residual add + RMSNorm (+bf16 output) -----------------
// D == 1024, block == 256, one block per row.
__global__ __launch_bounds__(256)
void add_rmsnorm_kernel(float* __restrict__ residual,
                        const float* __restrict__ hidden,
                        const float* __restrict__ w,
                        bf16_t* __restrict__ xout, int D, int doAdd) {
    __shared__ float red[256];
    int row = blockIdx.x, tid = threadIdx.x;
    float vals[4]; float ss = 0.f;
#pragma unroll
    for (int i = 0; i < 4; ++i) {
        int j = tid + i * 256;
        float h = hidden[(long)row * D + j];
        float r = doAdd ? (h + residual[(long)row * D + j]) : h;
        residual[(long)row * D + j] = r;
        vals[i] = r;
        ss += r * r;
    }
    red[tid] = ss;
    __syncthreads();
    for (int s = 128; s > 0; s >>= 1) {
        if (tid < s) red[tid] += red[tid + s];
        __syncthreads();
    }
    float scale = rsqrtf(red[0] * (1.0f / 1024.0f) + 1e-5f);
#pragma unroll
    for (int i = 0; i < 4; ++i) {
        int j = tid + i * 256;
        xout[(long)row * D + j] = f2bf(vals[i] * scale * w[j]);
    }
}

// ------------------------------ bf16 WMMA GEMM -----------------------------
// C[M,N] (f32) = A[M,K](bf16,row-major) * W[N,K](bf16,row-major)^T
// Block: 256 threads = 8 waves; each wave -> 16(M) x 64(N); block -> 128 x 64.
// K must be a multiple of 32; M a multiple of 16; N arbitrary (guarded).
__global__ __launch_bounds__(256)
void gemm_bf16_kernel(const bf16_t* __restrict__ A,
                      const bf16_t* __restrict__ W,
                      float* __restrict__ C, int M, int N, int K) {
    int lane = threadIdx.x & 31;
    int wave = threadIdx.x >> 5;
    int hl   = lane >> 4;      // 0 or 1
    int l16  = lane & 15;

    int m0 = blockIdx.y * 128 + wave * 16;
    if (m0 >= M) return;
    int nblk = blockIdx.x * 64;

    v8f acc0 = {}, acc1 = {}, acc2 = {}, acc3 = {};

    const bf16_t* Arow = A + (long)(m0 + l16) * K;

    int nr0 = nblk + l16, nr1 = nblk + 16 + l16;
    int nr2 = nblk + 32 + l16, nr3 = nblk + 48 + l16;
    bool g0 = nr0 < N, g1 = nr1 < N, g2 = nr2 < N, g3 = nr3 < N;
    const bf16_t* W0 = W + (long)(g0 ? nr0 : 0) * K;
    const bf16_t* W1 = W + (long)(g1 ? nr1 : 0) * K;
    const bf16_t* W2 = W + (long)(g2 ? nr2 : 0) * K;
    const bf16_t* W3 = W + (long)(g3 ? nr3 : 0) * K;

    for (int k0 = 0; k0 < K; k0 += 32) {
        // A fragment: 16x32 bf16 (contiguous K order: lane half selects K sub-block)
        BFrag a;
        a.h[0] = *(const v8bf*)(Arow + k0 + hl * 16);
        a.h[1] = *(const v8bf*)(Arow + k0 + hl * 16 + 8);

        int koff = k0 + hl * 16;   // B layout: lanes16-31 hold K=16..31
        BFrag b0, b1, b2, b3;
        b0.h[0] = *(const v8bf*)(W0 + koff); b0.h[1] = *(const v8bf*)(W0 + koff + 8);
        b1.h[0] = *(const v8bf*)(W1 + koff); b1.h[1] = *(const v8bf*)(W1 + koff + 8);
        b2.h[0] = *(const v8bf*)(W2 + koff); b2.h[1] = *(const v8bf*)(W2 + koff + 8);
        b3.h[0] = *(const v8bf*)(W3 + koff); b3.h[1] = *(const v8bf*)(W3 + koff + 8);

        acc0 = __builtin_amdgcn_wmma_f32_16x16x32_bf16(false, a.v, false, b0.v,
                                                       (short)0, acc0, false, false);
        acc1 = __builtin_amdgcn_wmma_f32_16x16x32_bf16(false, a.v, false, b1.v,
                                                       (short)0, acc1, false, false);
        acc2 = __builtin_amdgcn_wmma_f32_16x16x32_bf16(false, a.v, false, b2.v,
                                                       (short)0, acc2, false, false);
        acc3 = __builtin_amdgcn_wmma_f32_16x16x32_bf16(false, a.v, false, b3.v,
                                                       (short)0, acc3, false, false);
    }

    // C/D layout: lane -> col N=l16; VGPR v -> row M = v + 8*hl
    float* Crow = C + (long)m0 * N;
#pragma unroll
    for (int v = 0; v < 8; ++v) {
        long roff = (long)(v + 8 * hl) * N;
        if (g0) Crow[roff + nr0] = acc0[v];
        if (g1) Crow[roff + nr1] = acc1[v];
        if (g2) Crow[roff + nr2] = acc2[v];
        if (g3) Crow[roff + nr3] = acc3[v];
    }
}

// ------------------------- depthwise conv(4) + SiLU ------------------------
__global__ void conv_silu_kernel(const float* __restrict__ xz,
                                 const float* __restrict__ cw,
                                 const float* __restrict__ cb,
                                 float* __restrict__ uf,
                                 bf16_t* __restrict__ ub,
                                 int L, int Din) {
    long i = (long)blockIdx.x * blockDim.x + threadIdx.x;
    if (i >= (long)L * Din) return;
    int d = (int)(i % Din), l = (int)(i / Din);
    int stride2 = 2 * Din;                       // xz row stride (xp is low half)
    float acc = cb[d];
#pragma unroll
    for (int k = 0; k < 4; ++k) {
        int ls = l + k - 3;
        float xv = (ls >= 0) ? xz[(long)ls * stride2 + d] : 0.f;
        acc = fmaf(xv, cw[d * 4 + k], acc);
    }
    float u = silu_f(acc);
    uf[i] = u;
    ub[i] = f2bf(u);
}

// ----------------------- xdbl split: dt_r -> bf16 --------------------------
__global__ void split_dtr_kernel(const float* __restrict__ xdbl,
                                 bf16_t* __restrict__ dtr, int L) {
    long i = (long)blockIdx.x * blockDim.x + threadIdx.x;
    if (i >= (long)L * 64) return;
    int l = (int)(i / 64), r = (int)(i % 64);
    dtr[i] = f2bf(xdbl[(long)l * 96 + r]);
}

// ------------------------- softplus(dt + bias) (in place) ------------------
__global__ void softplus_kernel(float* __restrict__ dt,
                                const float* __restrict__ bias,
                                int L, int Din) {
    long i = (long)blockIdx.x * blockDim.x + threadIdx.x;
    if (i >= (long)L * Din) return;
    float x = dt[i] + bias[(int)(i % Din)];
    dt[i] = (x > 20.f) ? x : log1pf(__expf(x));
}

// ------------------------------ selective scan -----------------------------
// One thread per channel d; 16 states + 16 A coeffs in registers.
__global__ __launch_bounds__(256)
void scan_kernel(const float* __restrict__ dt, const float* __restrict__ u,
                 const float* __restrict__ xdbl,     // B at +64, C at +80, stride 96
                 const float* __restrict__ A_log, const float* __restrict__ Dv,
                 float* __restrict__ ys, int L, int Din) {
    int d = blockIdx.x * 256 + threadIdx.x;
    if (d >= Din) return;
    float A[16], s[16];
#pragma unroll
    for (int n = 0; n < 16; ++n) {
        A[n] = -__expf(A_log[(long)d * 16 + n]);
        s[n] = 0.f;
    }
    float Dd = Dv[d];
    for (int t = 0; t < L; ++t) {
        float dtv = dt[(long)t * Din + d];
        float uv  = u[(long)t * Din + d];
        float dtu = dtv * uv;
        const float* Bt = xdbl + (long)t * 96 + 64;
        const float* Ct = Bt + 16;
        float y = 0.f;
#pragma unroll
        for (int n = 0; n < 16; ++n) {
            s[n] = fmaf(s[n], __expf(dtv * A[n]), dtu * Bt[n]);
            y = fmaf(s[n], Ct[n], y);
        }
        ys[(long)t * Din + d] = fmaf(uv, Dd, y);
    }
}

// ----------------------------- y * silu(z) -> bf16 -------------------------
__global__ void gate_kernel(const float* __restrict__ ys,
                            const float* __restrict__ xz,
                            bf16_t* __restrict__ yg, int L, int Din) {
    long i = (long)blockIdx.x * blockDim.x + threadIdx.x;
    if (i >= (long)L * Din) return;
    int d = (int)(i % Din), l = (int)(i / Din);
    float z = xz[(long)l * (2 * Din) + Din + d];
    yg[i] = f2bf(ys[i] * silu_f(z));
}

// ---------------------------------------------------------------------------
extern "C" void kernel_launch(void* const* d_in, const int* in_sizes, int n_in,
                              void* d_out, int out_size, void* d_ws, size_t ws_size,
                              hipStream_t stream) {
    (void)in_sizes; (void)n_in; (void)out_size; (void)ws_size;

    const int   D_MODEL = 1024, D_INNER = 2048, VOCAB = 50280, L = 1024;
    const int   N_LAYER = 4, DT_RANK = 64, D_STATE = 16;

    const int*   ids       = (const int*)  d_in[0];
    const float* emb       = (const float*)d_in[1];
    const float* norm_w    = (const float*)d_in[2];
    const float* in_proj_w = (const float*)d_in[3];
    const float* conv_w    = (const float*)d_in[4];
    const float* conv_b    = (const float*)d_in[5];
    const float* x_proj_w  = (const float*)d_in[6];
    const float* dt_proj_w = (const float*)d_in[7];
    const float* dt_proj_b = (const float*)d_in[8];
    const float* A_log     = (const float*)d_in[9];
    const float* Dvec      = (const float*)d_in[10];
    const float* out_proj_w= (const float*)d_in[11];
    const float* norm_f_w  = (const float*)d_in[12];
    float* out = (float*)d_out;

    // --- workspace carving (256B aligned) ---
    char* p = (char*)d_ws;
    auto carve = [&](size_t bytes) -> void* {
        void* r = (void*)p;
        p += (bytes + 255) & ~(size_t)255;
        return r;
    };
    bf16_t* emb_bf  = (bf16_t*)carve((size_t)VOCAB * D_MODEL * 2);
    bf16_t* inw_bf  = (bf16_t*)carve((size_t)N_LAYER * 2 * D_INNER * D_MODEL * 2);
    bf16_t* outw_bf = (bf16_t*)carve((size_t)N_LAYER * D_MODEL * D_INNER * 2);
    bf16_t* xpw_bf  = (bf16_t*)carve((size_t)N_LAYER * (DT_RANK + 2 * D_STATE) * D_INNER * 2);
    bf16_t* dtw_bf  = (bf16_t*)carve((size_t)N_LAYER * D_INNER * DT_RANK * 2);
    float*  residual= (float*) carve((size_t)L * D_MODEL * 4);
    float*  hidden  = (float*) carve((size_t)L * D_MODEL * 4);
    bf16_t* x_bf    = (bf16_t*)carve((size_t)L * D_MODEL * 2);
    float*  xz      = (float*) carve((size_t)L * 2 * D_INNER * 4);
    float*  u_f     = (float*) carve((size_t)L * D_INNER * 4);
    bf16_t* u_bf    = (bf16_t*)carve((size_t)L * D_INNER * 2);
    float*  xdbl    = (float*) carve((size_t)L * 96 * 4);
    bf16_t* dtr_bf  = (bf16_t*)carve((size_t)L * DT_RANK * 2);
    float*  dtbuf   = (float*) carve((size_t)L * D_INNER * 4);
    float*  ys      = (float*) carve((size_t)L * D_INNER * 4);
    bf16_t* yg_bf   = (bf16_t*)carve((size_t)L * D_INNER * 2);
    bf16_t* hf_bf   = (bf16_t*)carve((size_t)L * D_MODEL * 2);

    auto cvt = [&](const float* s, bf16_t* d, long n) {
        int blocks = (int)((n + 255) / 256);
        if (blocks > 65535) blocks = 65535;
        f32_to_bf16_kernel<<<blocks, 256, 0, stream>>>(s, d, n);
    };
    auto gemm = [&](const bf16_t* A, const bf16_t* W, float* C, int M, int N, int K) {
        dim3 grid((N + 63) / 64, (M + 127) / 128);
        gemm_bf16_kernel<<<grid, 256, 0, stream>>>(A, W, C, M, N, K);
    };

    // --- one-time-per-call bf16 weight conversion (stays hot in 192MB L2) ---
    cvt(emb,        emb_bf,  (long)VOCAB * D_MODEL);
    cvt(in_proj_w,  inw_bf,  (long)N_LAYER * 2 * D_INNER * D_MODEL);
    cvt(out_proj_w, outw_bf, (long)N_LAYER * D_MODEL * D_INNER);
    cvt(x_proj_w,   xpw_bf,  (long)N_LAYER * 96 * D_INNER);
    cvt(dt_proj_w,  dtw_bf,  (long)N_LAYER * D_INNER * DT_RANK);

    // --- embedding gather ---
    {
        long n = (long)L * D_MODEL;
        embed_kernel<<<(int)((n + 255) / 256), 256, 0, stream>>>(ids, emb, hidden, L, D_MODEL);
    }

    long nLD  = (long)L * D_INNER;
    int  bLD  = (int)((nLD + 255) / 256);

    for (int i = 0; i < N_LAYER; ++i) {
        // residual (+=) hidden ; x = rmsnorm(residual) * norm_w  -> bf16
        add_rmsnorm_kernel<<<L, 256, 0, stream>>>(residual, hidden,
                                                  norm_w + (long)i * D_MODEL,
                                                  x_bf, D_MODEL, i > 0 ? 1 : 0);
        // xz = x @ in_proj_w^T   [1024 x 4096]
        gemm(x_bf, inw_bf + (long)i * 2 * D_INNER * D_MODEL, xz,
             L, 2 * D_INNER, D_MODEL);
        // u = silu(conv4(xp))
        conv_silu_kernel<<<bLD, 256, 0, stream>>>(xz,
            conv_w + (long)i * D_INNER * 4, conv_b + (long)i * D_INNER,
            u_f, u_bf, L, D_INNER);
        // xdbl = u @ x_proj_w^T  [1024 x 96]
        gemm(u_bf, xpw_bf + (long)i * 96 * D_INNER, xdbl, L, 96, D_INNER);
        // dt_r -> bf16
        {
            long n = (long)L * DT_RANK;
            split_dtr_kernel<<<(int)((n + 255) / 256), 256, 0, stream>>>(xdbl, dtr_bf, L);
        }
        // dt = softplus(dt_r @ dt_w^T + dt_b)   [1024 x 2048]
        gemm(dtr_bf, dtw_bf + (long)i * D_INNER * DT_RANK, dtbuf, L, D_INNER, DT_RANK);
        softplus_kernel<<<bLD, 256, 0, stream>>>(dtbuf,
            dt_proj_b + (long)i * D_INNER, L, D_INNER);
        // selective scan (sequential in L, per-channel registers)
        scan_kernel<<<D_INNER / 256, 256, 0, stream>>>(dtbuf, u_f, xdbl,
            A_log + (long)i * D_INNER * D_STATE, Dvec + (long)i * D_INNER,
            ys, L, D_INNER);
        // y * silu(z) -> bf16
        gate_kernel<<<bLD, 256, 0, stream>>>(ys, xz, yg_bf, L, D_INNER);
        // hidden = y @ out_proj_w^T   [1024 x 1024]
        gemm(yg_bf, outw_bf + (long)i * D_MODEL * D_INNER, hidden,
             L, D_MODEL, D_INNER);
    }

    // final: residual += hidden ; hf = rmsnorm(residual) * norm_f_w -> bf16
    add_rmsnorm_kernel<<<L, 256, 0, stream>>>(residual, hidden, norm_f_w,
                                              hf_bf, D_MODEL, 1);
    // logits = hf @ embedding^T   [1024 x 50280] -- dominant GEMM, B hot in L2
    gemm(hf_bf, emb_bf, out, L, VOCAB, D_MODEL);
}
